// SelfAttention_81003083203635
// MI455X (gfx1250) — matrix operand order repaired and checked by reading, compiled, not run on hardware
//
#include <hip/hip_runtime.h>
#include <math.h>

#define BB 8
#define CC 512
#define NN 1024
#define NHEADS 8
#define HD 64
#define NGROUPS 32
#define CPG 16  // CC / NGROUPS

#if defined(__has_builtin)
#if __has_builtin(__builtin_amdgcn_global_load_async_to_lds_b128) && \
    __has_builtin(__builtin_amdgcn_s_wait_asynccnt)
#define USE_ASYNC_LDS 1
#endif
#endif
#ifndef USE_ASYNC_LDS
#define USE_ASYNC_LDS 0
#endif

typedef __attribute__((ext_vector_type(16))) _Float16 v16h;
typedef __attribute__((ext_vector_type(8)))  _Float16 v8h;
typedef __attribute__((ext_vector_type(8)))  float    v8f;
typedef __attribute__((ext_vector_type(4)))  int      v4i;

static __device__ inline v8f zero8() {
  v8f z;
#pragma unroll
  for (int i = 0; i < 8; ++i) z[i] = 0.f;
  return z;
}

static __device__ inline v8f wmma_f16(v16h a, v16h b, v8f c) {
  // D = A(16x32 f16) x B(32x16 f16) + C(16x16 f32)
  return __builtin_amdgcn_wmma_f32_16x16x32_f16(false, a, false, b, (short)0, c,
                                                false, false);
}

// A-fragment (16x32, f16) from row-major [M, K] memory, leading dim ld (elems).
// Lane L<16: row L, K=0..15; lane L>=16: row L&15, K=16..31
// (contiguous order, matching load_b_frag).
static __device__ inline v16h load_a_frag(const _Float16* tile, int ld, int lane) {
  int row = lane & 15;
  int klo = (lane >> 4) << 4;  // 0 or 16
  const _Float16* p = tile + (size_t)row * ld + klo;
  v8h lo = *(const v8h*)(p);
  v8h hi = *(const v8h*)(p + 8);
  v16h a;
#pragma unroll
  for (int i = 0; i < 8; ++i) { a[i] = lo[i]; a[i + 8] = hi[i]; }
  return a;
}

// B-fragment (32x16, f16) from memory laid out [Ncols][K contiguous] with stride
// ld between columns. Lane L<16: col L, K=0..15; lane L>=16: col L&15, K=16..31.
static __device__ inline v16h load_b_frag(const _Float16* tile, int ld, int lane) {
  int col = lane & 15;
  int kb  = (lane >> 4) << 4;  // 0 or 16
  return *(const v16h*)(tile + (size_t)col * ld + kb);
}

#if USE_ASYNC_LDS
// 16-byte async global->LDS copy (ASYNCcnt-tracked, no VGPR round trip).
// ROCm signature: (v4i AS1* src, v4i AS3* dst, imm offset, imm cpol).
static __device__ inline void async_copy16(const _Float16* g, _Float16* l) {
  __builtin_amdgcn_global_load_async_to_lds_b128(
      (__attribute__((address_space(1))) v4i*)g,
      (__attribute__((address_space(3))) v4i*)l, 0, 0);
}
#endif

// ---------------------------------------------------------------- weights f32->f16
__global__ void cvt_kernel(const float* __restrict__ qw, const float* __restrict__ kw,
                           const float* __restrict__ vw, const float* __restrict__ ow,
                           _Float16* __restrict__ dst) {
  int i = blockIdx.x * 256 + threadIdx.x;  // 4 * 262144 total
  int m = i >> 18;
  int j = i & 262143;
  const float* src = (m == 0) ? qw : (m == 1) ? kw : (m == 2) ? vw : ow;
  dst[i] = (_Float16)src[j];
}

// ---------------------------------------------------------------- GroupNorm -> xn[B,N,C] f16
__global__ void gn_kernel(const float* __restrict__ x, const float* __restrict__ gw,
                          const float* __restrict__ gb, _Float16* __restrict__ xn) {
  const int b = blockIdx.x / NGROUPS;
  const int g = blockIdx.x % NGROUPS;
  const float* xp = x + ((size_t)b * CC + g * CPG) * NN;

  float s = 0.f, ss = 0.f;
  for (int i = threadIdx.x; i < CPG * NN; i += 256) {
    float v = xp[i];
    s += v; ss += v * v;
  }
#pragma unroll
  for (int m = 16; m >= 1; m >>= 1) { s += __shfl_xor(s, m); ss += __shfl_xor(ss, m); }
  __shared__ float red0[8], red1[8];
  int wave = threadIdx.x >> 5, lane = threadIdx.x & 31;
  if (lane == 0) { red0[wave] = s; red1[wave] = ss; }
  __syncthreads();
  if (threadIdx.x == 0) {
    float a = 0.f, c = 0.f;
    for (int i = 0; i < 8; ++i) { a += red0[i]; c += red1[i]; }
    red0[0] = a; red1[0] = c;
  }
  __syncthreads();
  const float inv_cnt = 1.f / (float)(CPG * NN);
  float mean = red0[0] * inv_cnt;
  float var  = red1[0] * inv_cnt - mean * mean;
  float rstd = rsqrtf(var + 1e-5f);

  for (int i = threadIdx.x; i < CPG * NN; i += 256) {
    int cl = i >> 10, n = i & (NN - 1);
    int c  = g * CPG + cl;
    float v = (xp[i] - mean) * rstd * gw[c] + gb[c];
    xn[((size_t)b * NN + n) * CC + c] = (_Float16)v;
  }
}

// ---------------------------------------------------------------- Q/K/V GEMMs
// which = blockIdx.z: 0 -> qT[B,N,C] (scaled), 1 -> kT[B,N,C], 2 -> v[B,C,N]
__global__ void qkv_kernel(const _Float16* __restrict__ xn,
                           const _Float16* __restrict__ wbuf,
                           const float* __restrict__ qb, const float* __restrict__ kb,
                           const float* __restrict__ vb,
                           _Float16* __restrict__ qT, _Float16* __restrict__ kT,
                           _Float16* __restrict__ vN) {
  const int which = blockIdx.z;
  const int b = blockIdx.y;
  const int otile = (blockIdx.x >> 4) * 128;
  const int ntile = (blockIdx.x & 15) * 64;
  const int wave = threadIdx.x >> 5, lane = threadIdx.x & 31;
  const int osub = otile + (wave >> 1) * 32;
  const int nsub = ntile + (wave & 1) * 32;
  const int hi = lane >> 4, col = lane & 15;

  const _Float16* W   = wbuf + (size_t)which * CC * CC;
  const float* bias   = (which == 0) ? qb : (which == 1) ? kb : vb;
  const float scale   = (which == 0) ? 0.125f : 1.0f;  // hd^-0.5

  v8f acc[2][2];
#pragma unroll
  for (int i = 0; i < 2; ++i)
#pragma unroll
    for (int j = 0; j < 2; ++j) acc[i][j] = zero8();

  const _Float16* xb = xn + ((size_t)b * NN + nsub) * CC;
  for (int k = 0; k < CC; k += 32) {
    if (k + 64 < CC) {
      __builtin_prefetch(xb + k + 64, 0, 1);
      __builtin_prefetch(W + (size_t)osub * CC + k + 64, 0, 1);
    }
    v16h a0 = load_a_frag(W + (size_t)osub * CC + k, CC, lane);
    v16h a1 = load_a_frag(W + (size_t)(osub + 16) * CC + k, CC, lane);
    v16h b0 = load_b_frag(xb + k, CC, lane);
    v16h b1 = load_b_frag(xb + (size_t)16 * CC + k, CC, lane);
    acc[0][0] = wmma_f16(a0, b0, acc[0][0]);
    acc[0][1] = wmma_f16(a0, b1, acc[0][1]);
    acc[1][0] = wmma_f16(a1, b0, acc[1][0]);
    acc[1][1] = wmma_f16(a1, b1, acc[1][1]);
  }

  if (which < 2) {
    _Float16* out = (which == 0) ? qT : kT;
#pragma unroll
    for (int i = 0; i < 2; ++i) {
      int o0 = osub + i * 16 + hi * 8;
#pragma unroll
      for (int j = 0; j < 2; ++j) {
        int n = nsub + j * 16 + col;
        v8h pk;
#pragma unroll
        for (int r = 0; r < 8; ++r)
          pk[r] = (_Float16)((acc[i][j][r] + bias[o0 + r]) * scale);
        *(v8h*)(out + ((size_t)b * NN + n) * CC + o0) = pk;  // transposed, contiguous
      }
    }
  } else {
#pragma unroll
    for (int i = 0; i < 2; ++i)
#pragma unroll
      for (int j = 0; j < 2; ++j) {
        int n = nsub + j * 16 + col;
#pragma unroll
        for (int r = 0; r < 8; ++r) {
          int o = osub + i * 16 + hi * 8 + r;
          vN[((size_t)b * CC + o) * NN + n] = (_Float16)(acc[i][j][r] + bias[o]);
        }
      }
  }
}

// ---------------------------------------------------------------- flash attention
// grid: (N/64, NHEADS, BB); block: 128 = 4 waves, each wave owns 16 query rows.
__global__ void attn_kernel(const _Float16* __restrict__ qT,
                            const _Float16* __restrict__ kT,
                            const _Float16* __restrict__ vN,
                            _Float16* __restrict__ oT) {
  const int b = blockIdx.z, h = blockIdx.y;
  const int wave = threadIdx.x >> 5, lane = threadIdx.x & 31;
  const int nbase = blockIdx.x * 64 + wave * 16;
  const int hi = lane >> 4, col = lane & 15;
  const int tid = threadIdx.x;
  (void)tid;

  __shared__ _Float16 Pbuf[4][16][64];  // per-wave 16x64 staging (8 KB)
#if USE_ASYNC_LDS
  __shared__ _Float16 Kt[64][64];  // token-major: [m][d]
  __shared__ _Float16 Vt[64][64];  // channel-major: [d][m]
#endif

  const _Float16* qp = qT + ((size_t)b * NN + nbase) * CC + h * HD;
  v16h Aq0 = load_a_frag(qp, CC, lane);        // d = 0..31
  v16h Aq1 = load_a_frag(qp + 32, CC, lane);   // d = 32..63

  v8f O[4];
#pragma unroll
  for (int d = 0; d < 4; ++d) O[d] = zero8();
  float Mrow[8], Lrow[8];
#pragma unroll
  for (int r = 0; r < 8; ++r) { Mrow[r] = -1e30f; Lrow[r] = 0.f; }

  for (int mb = 0; mb < NN; mb += 64) {
#if USE_ASYNC_LDS
    // cooperative async stage: 64x64 K tile + 64x64 V tile, 16B chunks.
#pragma unroll
    for (int i = 0; i < 4; ++i) {
      int c = tid + 128 * i;          // 512 chunks per tile
      int row = c >> 3, cc = (c & 7) * 8;
      async_copy16(kT + ((size_t)b * NN + mb + row) * CC + h * HD + cc,
                   &Kt[row][cc]);
      async_copy16(vN + ((size_t)b * CC + h * HD + row) * NN + mb + cc,
                   &Vt[row][cc]);
    }
    __builtin_amdgcn_s_wait_asynccnt(0);
    __syncthreads();
#endif
    v8f S[4];
#pragma unroll
    for (int mt = 0; mt < 4; ++mt) {
      v8f s = zero8();
#if USE_ASYNC_LDS
      const _Float16* kp = &Kt[mt * 16][0];
      s = wmma_f16(Aq0, load_b_frag(kp, 64, lane), s);
      s = wmma_f16(Aq1, load_b_frag(kp + 32, 64, lane), s);
#else
      const _Float16* kp = kT + ((size_t)b * NN + mb + mt * 16) * CC + h * HD;
      s = wmma_f16(Aq0, load_b_frag(kp, CC, lane), s);
      s = wmma_f16(Aq1, load_b_frag(kp + 32, CC, lane), s);
#endif
      S[mt] = s;
    }
    // online softmax: rows live in (VGPR r, half-wave); columns across 16 lanes
#pragma unroll
    for (int r = 0; r < 8; ++r) {
      float mx = fmaxf(fmaxf(S[0][r], S[1][r]), fmaxf(S[2][r], S[3][r]));
#pragma unroll
      for (int m = 8; m >= 1; m >>= 1) mx = fmaxf(mx, __shfl_xor(mx, m));
      float newM = fmaxf(Mrow[r], mx);
      float corr = __expf(Mrow[r] - newM);
      Mrow[r] = newM;
      float rsum = 0.f;
#pragma unroll
      for (int mt = 0; mt < 4; ++mt) {
        float p = __expf(S[mt][r] - newM);
        S[mt][r] = p;
        rsum += p;
      }
#pragma unroll
      for (int m = 8; m >= 1; m >>= 1) rsum += __shfl_xor(rsum, m);
      Lrow[r] = Lrow[r] * corr + rsum;
#pragma unroll
      for (int dt = 0; dt < 4; ++dt) O[dt][r] *= corr;
    }
    // C-layout -> A-layout via LDS
#pragma unroll
    for (int mt = 0; mt < 4; ++mt)
#pragma unroll
      for (int r = 0; r < 8; ++r)
        Pbuf[wave][r + hi * 8][mt * 16 + col] = (_Float16)S[mt][r];
    __syncthreads();
    v16h P0 = load_a_frag(&Pbuf[wave][0][0], 64, lane);       // m = 0..31
    v16h P1 = load_a_frag(&Pbuf[wave][0][0] + 32, 64, lane);  // m = 32..63
#pragma unroll
    for (int dt = 0; dt < 4; ++dt) {
#if USE_ASYNC_LDS
      const _Float16* vp = &Vt[dt * 16][0];
      O[dt] = wmma_f16(P0, load_b_frag(vp, 64, lane), O[dt]);
      O[dt] = wmma_f16(P1, load_b_frag(vp + 32, 64, lane), O[dt]);
#else
      const _Float16* vp = vN + ((size_t)b * CC + h * HD + dt * 16) * NN + mb;
      O[dt] = wmma_f16(P0, load_b_frag(vp, NN, lane), O[dt]);
      O[dt] = wmma_f16(P1, load_b_frag(vp + 32, NN, lane), O[dt]);
#endif
    }
    __syncthreads();
  }
  // epilogue: stage O through LDS so the token-major store is coalesced 16B ops
#pragma unroll
  for (int dt = 0; dt < 4; ++dt)
#pragma unroll
    for (int r = 0; r < 8; ++r)
      Pbuf[wave][r + hi * 8][dt * 16 + col] = (_Float16)(O[dt][r] / Lrow[r]);
  __syncthreads();
#pragma unroll
  for (int i = 0; i < 4; ++i) {
    int c = lane + 32 * i;  // 128 16B-chunks per wave tile
    int row = c >> 3, cc = (c & 7) * 8;
    v8h pk = *(const v8h*)(&Pbuf[wave][row][cc]);
    *(v8h*)(oT + ((size_t)b * NN + nbase + row) * CC + h * HD + cc) = pk;
  }
}

// ---------------------------------------------------------------- out-proj + residual
__global__ void proj_kernel(const _Float16* __restrict__ oT,
                            const _Float16* __restrict__ wo,
                            const float* __restrict__ ob, const float* __restrict__ x,
                            float* __restrict__ out) {
  const int b = blockIdx.y;
  const int otile = (blockIdx.x >> 4) * 128;
  const int ntile = (blockIdx.x & 15) * 64;
  const int wave = threadIdx.x >> 5, lane = threadIdx.x & 31;
  const int osub = otile + (wave >> 1) * 32;
  const int nsub = ntile + (wave & 1) * 32;
  const int hi = lane >> 4, col = lane & 15;

  v8f acc[2][2];
#pragma unroll
  for (int i = 0; i < 2; ++i)
#pragma unroll
    for (int j = 0; j < 2; ++j) acc[i][j] = zero8();

  const _Float16* xb = oT + ((size_t)b * NN + nsub) * CC;
  for (int k = 0; k < CC; k += 32) {
    if (k + 64 < CC) {
      __builtin_prefetch(xb + k + 64, 0, 1);
      __builtin_prefetch(wo + (size_t)osub * CC + k + 64, 0, 1);
    }
    v16h a0 = load_a_frag(wo + (size_t)osub * CC + k, CC, lane);
    v16h a1 = load_a_frag(wo + (size_t)(osub + 16) * CC + k, CC, lane);
    v16h b0 = load_b_frag(xb + k, CC, lane);
    v16h b1 = load_b_frag(xb + (size_t)16 * CC + k, CC, lane);
    acc[0][0] = wmma_f16(a0, b0, acc[0][0]);
    acc[0][1] = wmma_f16(a0, b1, acc[0][1]);
    acc[1][0] = wmma_f16(a1, b0, acc[1][0]);
    acc[1][1] = wmma_f16(a1, b1, acc[1][1]);
  }
#pragma unroll
  for (int i = 0; i < 2; ++i)
#pragma unroll
    for (int j = 0; j < 2; ++j) {
      int n = nsub + j * 16 + col;
#pragma unroll
      for (int r = 0; r < 8; ++r) {
        int o = osub + i * 16 + hi * 8 + r;
        size_t idx = ((size_t)b * CC + o) * NN + n;
        out[idx] = acc[i][j][r] + ob[o] + x[idx];
      }
    }
}

extern "C" void kernel_launch(void* const* d_in, const int* in_sizes, int n_in,
                              void* d_out, int out_size, void* d_ws, size_t ws_size,
                              hipStream_t stream) {
  const float* x   = (const float*)d_in[0];
  const float* gnw = (const float*)d_in[1];
  const float* gnb = (const float*)d_in[2];
  const float* qw  = (const float*)d_in[3];
  const float* qb  = (const float*)d_in[4];
  const float* kw  = (const float*)d_in[5];
  const float* kb  = (const float*)d_in[6];
  const float* vw  = (const float*)d_in[7];
  const float* vb  = (const float*)d_in[8];
  const float* ow  = (const float*)d_in[9];
  const float* ob  = (const float*)d_in[10];
  float* out = (float*)d_out;

  _Float16* ws = (_Float16*)d_ws;
  const size_t WSZ = (size_t)CC * CC;       // 262144 elems per weight matrix
  const size_t ASZ = (size_t)BB * NN * CC;  // 4 Mi elems per activation tensor
  _Float16* wbuf = ws;              // 4 weight matrices, f16
  _Float16* xn   = wbuf + 4 * WSZ;  // [B,N,C]
  _Float16* qT   = xn + ASZ;        // [B,N,C] (pre-scaled)
  _Float16* kT   = qT + ASZ;        // [B,N,C]
  _Float16* vN   = kT + ASZ;        // [B,C,N]
  _Float16* oTb  = vN + ASZ;        // [B,N,C]

  cvt_kernel<<<dim3((unsigned)((4 * WSZ) / 256)), dim3(256), 0, stream>>>(
      qw, kw, vw, ow, wbuf);
  gn_kernel<<<dim3(BB * NGROUPS), dim3(256), 0, stream>>>(x, gnw, gnb, xn);
  qkv_kernel<<<dim3(64, BB, 3), dim3(256), 0, stream>>>(xn, wbuf, qb, kb, vb, qT,
                                                        kT, vN);
  attn_kernel<<<dim3(NN / 64, NHEADS, BB), dim3(128), 0, stream>>>(qT, kT, vN, oTb);
  proj_kernel<<<dim3(64, BB), dim3(256), 0, stream>>>(oTb, wbuf + 3 * WSZ, ob, x,
                                                      out);
}
